// DTD_54468775248538
// MI455X (gfx1250) — matrix operand order repaired and checked by reading, compiled, not run on hardware
//
#include <hip/hip_runtime.h>
#include <hip/hip_bf16.h>

#define B_   128
#define T_   32
#define C_   512
#define HH_  256
#define HW_  256
#define NC_  6625
#define NCP_ 6640   // NC padded to multiple of 16

typedef __attribute__((ext_vector_type(2)))  float        v2f;
typedef __attribute__((ext_vector_type(8)))  float        v8f;
typedef __attribute__((ext_vector_type(16))) __bf16       v16bf;
typedef __attribute__((ext_vector_type(4)))  unsigned int v4u;
typedef __attribute__((ext_vector_type(8)))  int          v8i;
typedef __attribute__((ext_vector_type(4)))  int          v4i;

__device__ __forceinline__ v8f wmma_f32x4(v2f a, v2f b, v8f c) {
  return __builtin_amdgcn_wmma_f32_16x16x4_f32(false, a, false, b, (short)0, c, false, false);
}
__device__ __forceinline__ v8f wmma_bf16x32(v16bf a, v16bf b, v8f c) {
  return __builtin_amdgcn_wmma_f32_16x16x32_bf16(false, a, false, b, (short)0, c, false, false);
}
__device__ __forceinline__ float sigmoidf_(float x) { return 1.0f / (1.0f + __expf(-x)); }

__device__ __forceinline__ unsigned short f2bf(float f) {
  unsigned int u = __float_as_uint(f);
  u += 0x7FFFu + ((u >> 16) & 1u);          // round-to-nearest-even
  return (unsigned short)(u >> 16);
}

// ---------------------------------------------------------------------------
// 1) attention normalization: att_n[b*T+t, :] = att / sum_hw(att)
//    also writes the out_attentions section of d_out.
// ---------------------------------------------------------------------------
__global__ void att_norm_kernel(const float* __restrict__ att,
                                float* __restrict__ att_n,
                                float* __restrict__ out_att) {
  __shared__ float red[8];
  int row = blockIdx.x;                 // b*T + t  (0..4095)
  int tid = threadIdx.x;                // 0..255 == hw
  float v = att[(long)row * HW_ + tid];
  float s = v;
  #pragma unroll
  for (int o = 16; o > 0; o >>= 1) s += __shfl_xor(s, o, 32);
  if ((tid & 31) == 0) red[tid >> 5] = s;
  __syncthreads();
  float tot = red[0] + red[1] + red[2] + red[3] + red[4] + red[5] + red[6] + red[7];
  float r = v * (1.0f / tot);
  att_n[(long)row * HW_ + tid]  = r;
  out_att[(long)row * HW_ + tid] = r;
}

// ---------------------------------------------------------------------------
// 2) Generic NT f32 WMMA GEMM: C[m,n] = sum_k A[m,k]*B[n,k] + bias(+bias2)
//    optional batch dim (blockIdx.y) with per-operand batch strides, and
//    strided output addressing (lets us write cseq in [T,B,C] layout).
//    Requires tiles_m*tiles_n waves exactly (grid.x = tiles/8, blockDim=256).
// ---------------------------------------------------------------------------
__global__ void wmma_gemm_nt_f32(const float* __restrict__ A, int lda, long a_bs,
                                 const float* __restrict__ Bw, int ldb, long b_bs,
                                 const float* __restrict__ bias,
                                 const float* __restrict__ bias2,
                                 float* __restrict__ Cout, long c_sm, long c_sn, long c_bs,
                                 int tiles_m, int K) {
  int wave = blockIdx.x * 8 + (threadIdx.x >> 5);
  int lane = threadIdx.x & 31;
  int half = lane >> 4;
  int ml   = lane & 15;
  int mt = wave % tiles_m;
  int nt = wave / tiles_m;
  long ab = (long)blockIdx.y * a_bs;
  long bb = (long)blockIdx.y * b_bs;
  long cb = (long)blockIdx.y * c_bs;
  int m = mt * 16 + ml;
  int n = nt * 16 + ml;
  float bv = (bias ? bias[n] : 0.0f) + (bias2 ? bias2[n] : 0.0f);
  v8f acc;
  #pragma unroll
  for (int r = 0; r < 8; ++r) acc[r] = bv;
  const float* ap = A + ab + (long)m * lda;
  const float* bp = Bw + bb + (long)n * ldb;
  for (int k = 0; k < K; k += 4) {
    v2f a = *(const v2f*)(ap + k + 2 * half);   // A 16x4: lanes<16 K=0,1; lanes>=16 K=2,3
    v2f b = *(const v2f*)(bp + k + 2 * half);   // B 4x16 mirrored
    acc = wmma_f32x4(a, b, acc);
  }
  #pragma unroll
  for (int r = 0; r < 8; ++r) {
    int mm = mt * 16 + r + 8 * half;            // C/D layout: M = r + 8*(lane/16)
    Cout[cb + (long)mm * c_sm + (long)n * c_sn] = acc[r];
  }
}

// ---------------------------------------------------------------------------
// 3) LSTM recurrent step (both directions). Each wave owns one
//    (16-batch x 16-hidden) tile and computes all 4 gate GEMM tiles via
//    f32 WMMA, then the gate nonlinearity + c/h update in registers.
//    128 waves/dir: mt in [0,8), ut in [0,16). grid = (16, 2dirs), block 256.
// ---------------------------------------------------------------------------
__global__ void lstm_step_kernel(const float* __restrict__ xg_f, const float* __restrict__ xg_b,
                                 const float* __restrict__ whh_f, const float* __restrict__ whh_b,
                                 const float* __restrict__ h_in_base, float* __restrict__ h_out_base,
                                 float* __restrict__ c_base, float* __restrict__ cseq_bi, int t) {
  int dir  = blockIdx.y;
  int wave = blockIdx.x * 8 + (threadIdx.x >> 5);   // 0..127
  int lane = threadIdx.x & 31, half = lane >> 4, ml = lane & 15;
  int mt = wave & 7;          // batch tile
  int ut = wave >> 3;         // hidden-unit tile
  int t_eff = (dir == 0) ? t : (T_ - 1 - t);
  const float* xg   = (dir == 0 ? xg_f : xg_b) + (long)t_eff * B_ * (4 * HH_);
  const float* whh  = (dir == 0 ? whh_f : whh_b);
  const float* h_in = h_in_base + dir * (B_ * HH_);
  float* h_out      = h_out_base + dir * (B_ * HH_);
  float* c_st       = c_base + dir * (B_ * HH_);
  int u = ut * 16 + ml;

  v8f acc[4];   // gates i, f, g, o — seeded with precomputed x@Wih + biases
  #pragma unroll
  for (int g = 0; g < 4; ++g)
    #pragma unroll
    for (int r = 0; r < 8; ++r) {
      int mb = mt * 16 + r + 8 * half;
      acc[g][r] = xg[(long)mb * (4 * HH_) + g * HH_ + u];
    }

  const float* ap = h_in + (long)(mt * 16 + ml) * HH_;
  for (int k = 0; k < HH_; k += 4) {
    v2f a = *(const v2f*)(ap + k + 2 * half);
    #pragma unroll
    for (int g = 0; g < 4; ++g) {
      v2f b = *(const v2f*)(whh + (long)(g * HH_ + u) * HH_ + k + 2 * half);
      acc[g] = wmma_f32x4(a, b, acc[g]);
    }
  }

  #pragma unroll
  for (int r = 0; r < 8; ++r) {
    int mb = mt * 16 + r + 8 * half;
    float ig = sigmoidf_(acc[0][r]);
    float fg = sigmoidf_(acc[1][r]);
    float gg = tanhf(acc[2][r]);
    float og = sigmoidf_(acc[3][r]);
    float c_old = c_st[(long)mb * HH_ + u];
    float c_new = fg * c_old + ig * gg;
    float h     = og * tanhf(c_new);
    c_st[(long)mb * HH_ + u]  = c_new;
    h_out[(long)mb * HH_ + u] = h;
    cseq_bi[(long)t_eff * B_ * C_ + (long)mb * C_ + dir * HH_ + u] = h;
  }
}

// ---------------------------------------------------------------------------
// 4) build GRU input x = concat(cseq_bi[t,b,:], char_emb[prev_idx[b,t]])
// ---------------------------------------------------------------------------
__global__ void gru_gather_kernel(const float* __restrict__ cseq_bi,
                                  const int* __restrict__ text,
                                  const float* __restrict__ emb,
                                  float* __restrict__ x) {
  long idx = (long)blockIdx.x * blockDim.x + threadIdx.x;   // < 4096*1024
  int  col = (int)(idx & 1023);
  long row = idx >> 10;                                     // t*B + b
  float v;
  if (col < C_) {
    v = cseq_bi[row * C_ + col];
  } else {
    int b = (int)(row & 127);
    int t = (int)(row >> 7);
    int pid = (t == 0) ? 0 : text[b * T_ + (t - 1)];
    v = emb[(long)pid * C_ + (col - C_)];
  }
  x[idx] = v;
}

// ---------------------------------------------------------------------------
// 5) GRU recurrent step. 256 waves: mt in [0,8), ut in [0,32). grid=32,block=256.
// ---------------------------------------------------------------------------
__global__ void gru_step_kernel(const float* __restrict__ gi,
                                const float* __restrict__ whh,
                                const float* __restrict__ b_hh,
                                const float* __restrict__ h_in, float* __restrict__ h_out,
                                float* __restrict__ gru_res, int t) {
  int wave = blockIdx.x * 8 + (threadIdx.x >> 5);   // 0..255
  int lane = threadIdx.x & 31, half = lane >> 4, ml = lane & 15;
  int mt = wave & 7;
  int ut = wave >> 3;
  int u = ut * 16 + ml;

  v8f acc[3];   // hr, hz, hn — seeded with b_hh
  #pragma unroll
  for (int g = 0; g < 3; ++g) {
    float bv = b_hh[g * C_ + u];
    #pragma unroll
    for (int r = 0; r < 8; ++r) acc[g][r] = bv;
  }

  const float* ap = h_in + (long)(mt * 16 + ml) * C_;
  for (int k = 0; k < C_; k += 4) {
    v2f a = *(const v2f*)(ap + k + 2 * half);
    #pragma unroll
    for (int g = 0; g < 3; ++g) {
      v2f b = *(const v2f*)(whh + (long)(g * C_ + u) * C_ + k + 2 * half);
      acc[g] = wmma_f32x4(a, b, acc[g]);
    }
  }

  const float* git = gi + (long)t * B_ * (3 * C_);
  #pragma unroll
  for (int r = 0; r < 8; ++r) {
    int mb = mt * 16 + r + 8 * half;
    long gb = (long)mb * (3 * C_);
    float ir  = git[gb + u];
    float iz  = git[gb + C_ + u];
    float inn = git[gb + 2 * C_ + u];
    float rg = sigmoidf_(ir + acc[0][r]);
    float zg = sigmoidf_(iz + acc[1][r]);
    float ng = tanhf(inn + rg * acc[2][r]);
    float ho = h_in[(long)mb * C_ + u];
    float hn = (1.0f - zg) * ng + zg * ho;
    h_out[(long)mb * C_ + u] = hn;
    gru_res[(long)t * B_ * C_ + (long)mb * C_ + u] = hn;
  }
}

// ---------------------------------------------------------------------------
// 6) f32 -> bf16 conversion (generator operands). gen_w padded to NCP_ rows.
// ---------------------------------------------------------------------------
__global__ void conv_bf16_kernel(const float* __restrict__ src,
                                 unsigned short* __restrict__ dst, long n) {
  long i = (long)blockIdx.x * blockDim.x + threadIdx.x;
  if (i < n) dst[i] = f2bf(src[i]);
}
__global__ void conv_genw_kernel(const float* __restrict__ w,
                                 unsigned short* __restrict__ dst) {
  long i = (long)blockIdx.x * blockDim.x + threadIdx.x;   // < NCP_*512
  long row = i >> 9;
  int  col = (int)(i & 511);
  dst[i] = (row < NC_) ? f2bf(w[row * C_ + col]) : (unsigned short)0;
}

// ---------------------------------------------------------------------------
// 7) Generator GEMM, bf16 WMMA (f32 accumulate) + bias.
//    All 8 waves of a block share one N-tile (nt = wave/256, tiles_m=256
//    divisible by 8), so ONE Tensor-Data-Mover DMA per block stages the
//    16x512 bf16 weight tile (16 KB) into LDS. D# descriptor built per
//    CDNA5 ISA (2-D tensor: trailing groups zero).
//    B-fragments then come from LDS (ds_load_b128), A-fragments via
//    global b128 loads. Output rows remapped t*B+b -> b*T+t.
// ---------------------------------------------------------------------------
__global__ void gen_gemm_kernel(const unsigned short* __restrict__ Abf,   // [4096,512]
                                const unsigned short* __restrict__ Wbf,   // [NCP_,512]
                                const float* __restrict__ bias,           // [NC_]
                                float* __restrict__ out) {                // [B*T, NC_]
  __shared__ unsigned short btile[16 * 512];   // 16 KB weight tile, LDS offset 0
  int wid  = threadIdx.x >> 5;
  int wave = blockIdx.x * 8 + wid;
  int lane = threadIdx.x & 31, half = lane >> 4, ml = lane & 15;
  const int tiles_m = 256;
  int mt = wave % tiles_m;
  int nt = wave / tiles_m;          // identical for all 8 waves of this block

  if (wid == 0) {
    // TDM: DMA Wbf[nt*16 .. nt*16+16) x [0,512) (bf16) into LDS.
    unsigned long long ga =
        (unsigned long long)(const char*)(Wbf + (size_t)nt * 16 * C_);
    v4u g0;
    g0[0] = 1u;                                             // count=1, user mode
    g0[1] = 0u;                                             // lds_addr = 0 (btile)
    g0[2] = (unsigned int)ga;                               // global_addr[31:0]
    g0[3] = ((unsigned int)(ga >> 32) & 0x01FFFFFFu)        // global_addr[56:32]
            | 0x80000000u;                                  // type=2 ("image")
    v8i g1;
    g1[0] = (int)(1u << 16);        // workgroup_mask=0, data_size=1 (2 bytes)
    g1[1] = (int)(512u << 16);      // tensor_dim0 = 512 (bits 79:48, low half)
    g1[2] = (int)(6640u << 16);     // tensor_dim0 hi=0 | tensor_dim1 = 6640 (lo)
    g1[3] = (int)(512u << 16);      // tensor_dim1 hi=0 | tile_dim0 = 512
    g1[4] = 16;                     // tile_dim1 = 16, tile_dim2 = 0
    g1[5] = 512;                    // tensor_dim0_stride = 512 (low 32)
    g1[6] = 0;                      // stride hi, tensor_dim1_stride lo
    g1[7] = 0;
    v4i gz4 = {0, 0, 0, 0};         // groups 2/3: 2-D tensor -> zero
    v8i gz8 = {0, 0, 0, 0, 0, 0, 0, 0};
    // amdgpu-toolchain (clang-23) 6-arg form:
    // (uint32x4 g0, int32x8 g1, int32x4, int32x4, int32x8, i32 cpol)
    __builtin_amdgcn_tensor_load_to_lds(g0, g1, gz4, gz4, gz8, 0);
    __builtin_amdgcn_s_wait_tensorcnt(0);                   // s_wait_tensorcnt 0
  }
  __syncthreads();

  int mrow = mt * 16 + ml;
  int ncol = nt * 16 + ml;
  float bv = (ncol < NC_) ? bias[ncol] : 0.0f;
  v8f acc;
  #pragma unroll
  for (int r = 0; r < 8; ++r) acc[r] = bv;

  const unsigned short* ap = Abf + (long)mrow * C_;
  for (int k = 0; k < C_; k += 32) {
    __builtin_prefetch(ap + k + 64, 0, 1);   // global_prefetch_b8
    union { v16bf v; v4u q[2]; } ua, ub;
    // A 16x32 bf16: lanes<16 hold K=[0,8)+[16,24); lanes>=16 hold K=[8,16)+[24,32)
    ua.q[0] = *(const v4u*)(ap + k + 8 * half);
    ua.q[1] = *(const v4u*)(ap + k + 16 + 8 * half);
    // B 32x16 bf16 from LDS: col = lane%16, K=[8*half,+8)+[16+8*half,+8) (matches A)
    const unsigned short* bl = btile + ml * C_ + k + 8 * half;
    ub.q[0] = *(const v4u*)(bl);
    ub.q[1] = *(const v4u*)(bl + 16);
    acc = wmma_bf16x32(ua.v, ub.v, acc);
  }

  if (ncol < NC_) {
    #pragma unroll
    for (int r = 0; r < 8; ++r) {
      int m  = mt * 16 + r + 8 * half;     // m = t*B + b
      int tt = m >> 7;
      int bb = m & 127;
      out[(long)(bb * T_ + tt) * NC_ + ncol] = acc[r];
    }
  }
}

// ---------------------------------------------------------------------------
extern "C" void kernel_launch(void* const* d_in, const int* in_sizes, int n_in,
                              void* d_out, int out_size, void* d_ws, size_t ws_size,
                              hipStream_t stream) {
  (void)in_sizes; (void)n_in; (void)out_size; (void)ws_size;
  const float* feature  = (const float*)d_in[0];
  const float* attmap   = (const float*)d_in[1];
  const int*   text     = (const int*)d_in[2];
  const float* emb      = (const float*)d_in[4];
  const float* w_ih_f   = (const float*)d_in[5];
  const float* w_hh_f   = (const float*)d_in[6];
  const float* b_ih_f   = (const float*)d_in[7];
  const float* b_hh_f   = (const float*)d_in[8];
  const float* w_ih_b   = (const float*)d_in[9];
  const float* w_hh_b   = (const float*)d_in[10];
  const float* b_ih_b   = (const float*)d_in[11];
  const float* b_hh_b   = (const float*)d_in[12];
  const float* gru_w_ih = (const float*)d_in[13];
  const float* gru_w_hh = (const float*)d_in[14];
  const float* gru_b_ih = (const float*)d_in[15];
  const float* gru_b_hh = (const float*)d_in[16];
  const float* gen_w    = (const float*)d_in[17];
  const float* gen_b    = (const float*)d_in[18];
  float* out = (float*)d_out;

  char* ws = (char*)d_ws;
  size_t off = 0;
  auto alloc = [&](size_t bytes) -> char* {
    char* p = ws + off;
    off += (bytes + 255) & ~(size_t)255;
    return p;
  };
  float* att_n   = (float*)alloc(4096ull * 256 * 4);
  float* cseq    = (float*)alloc(4096ull * 512 * 4);
  float* xg_f    = (float*)alloc(4096ull * 1024 * 4);
  float* xg_b    = (float*)alloc(4096ull * 1024 * 4);
  float* cseq_bi = (float*)alloc(4096ull * 512 * 4);
  float* gru_x   = (float*)alloc(4096ull * 1024 * 4);
  float* gi      = (float*)alloc(4096ull * 1536 * 4);
  float* gru_res = (float*)alloc(4096ull * 512 * 4);
  // state block (zeroed once per call): lstm_h[2 parity][2 dir][128*256],
  // lstm_c[2 dir][128*256], gru_h[2 parity][128*512]
  const size_t state_elems = 4 * 32768 + 2 * 32768 + 2 * 65536;
  float* states = (float*)alloc(state_elems * 4);
  float* lstm_h = states;
  float* lstm_c = states + 4 * 32768;
  float* gru_h  = states + 6 * 32768;
  unsigned short* a_bf = (unsigned short*)alloc(4096ull * 512 * 2);
  unsigned short* w_bf = (unsigned short*)alloc((size_t)NCP_ * 512 * 2);

  (void)hipMemsetAsync(states, 0, state_elems * sizeof(float), stream);

  // 1) attention normalization (+ write out_attentions section of d_out)
  const long OUT_RES_ELEMS = (long)B_ * T_ * NC_;   // 27,136,000
  att_norm_kernel<<<4096, 256, 0, stream>>>(attmap, att_n, out + OUT_RES_ELEMS);

  // 2) attention pooling: per-batch NT GEMM [C x HW]x[T x HW]^T -> cseq [T,B,C]
  wmma_gemm_nt_f32<<<dim3(8, 128), 256, 0, stream>>>(
      feature, HW_, (long)C_ * HW_,
      att_n,   HW_, (long)T_ * HW_,
      nullptr, nullptr,
      cseq, /*c_sm=*/1, /*c_sn=*/(long)B_ * C_, /*c_bs=*/(long)C_,
      /*tiles_m=*/C_ / 16, /*K=*/HW_);

  // 3) hoisted LSTM input GEMMs: xg = cseq @ Wih^T + (b_ih + b_hh)
  wmma_gemm_nt_f32<<<2048, 256, 0, stream>>>(cseq, C_, 0, w_ih_f, C_, 0,
      b_ih_f, b_hh_f, xg_f, 4 * HH_, 1, 0, 256, C_);
  wmma_gemm_nt_f32<<<2048, 256, 0, stream>>>(cseq, C_, 0, w_ih_b, C_, 0,
      b_ih_b, b_hh_b, xg_b, 4 * HH_, 1, 0, 256, C_);

  // 4) LSTM recurrence, both directions per launch, ping-pong h
  for (int t = 0; t < T_; ++t) {
    const float* h_in = lstm_h + (t & 1) * 2 * (B_ * HH_);
    float* h_out      = lstm_h + ((t + 1) & 1) * 2 * (B_ * HH_);
    lstm_step_kernel<<<dim3(16, 2), 256, 0, stream>>>(
        xg_f, xg_b, w_hh_f, w_hh_b, h_in, h_out, lstm_c, cseq_bi, t);
  }

  // 5) GRU input gather (concat LSTM output with prev-char embeddings)
  gru_gather_kernel<<<16384, 256, 0, stream>>>(cseq_bi, text, emb, gru_x);

  // 6) hoisted GRU input GEMM: gi = x @ Wih^T + b_ih
  wmma_gemm_nt_f32<<<3072, 256, 0, stream>>>(gru_x, 2 * C_, 0, gru_w_ih, 2 * C_, 0,
      gru_b_ih, nullptr, gi, 3 * C_, 1, 0, 256, 2 * C_);

  // 7) GRU recurrence, ping-pong h
  for (int t = 0; t < T_; ++t) {
    const float* h_in = gru_h + (t & 1) * (B_ * C_);
    float* h_out      = gru_h + ((t + 1) & 1) * (B_ * C_);
    gru_step_kernel<<<32, 256, 0, stream>>>(gi, gru_w_hh, gru_b_hh, h_in, h_out, gru_res, t);
  }

  // 8) bf16 conversion for the generator GEMM
  conv_bf16_kernel<<<(4096 * 512) / 256, 256, 0, stream>>>(gru_res, a_bf, 4096ll * 512);
  conv_genw_kernel<<<(NCP_ * 512) / 256, 256, 0, stream>>>(gen_w, w_bf);

  // 9) generator: [4096,512] x [NC,512]^T + gen_b (TDM-staged weights)
  gen_gemm_kernel<<<(256 * (NCP_ / 16)) / 8, 256, 0, stream>>>(a_bf, w_bf, gen_b, out);
}
